// ContextAttention_13657996001458
// MI455X (gfx1250) — matrix operand order repaired and checked by reading, not compiled, not run on hardware
//
#include <hip/hip_runtime.h>
#include <hip/hip_bf16.h>
#include <stdint.h>

typedef _Float16 v16h  __attribute__((ext_vector_type(16)));
typedef _Float16 half8 __attribute__((ext_vector_type(8)));
typedef float    v8f   __attribute__((ext_vector_type(8)));

#define BB 4
#define SS 4096
#define DD 512

#define SHUF16(lo, hi) __builtin_shufflevector((lo), (hi), 0,1,2,3,4,5,6,7,8,9,10,11,12,13,14,15)

// ---------------------------------------------------------------------------
// f32 -> f16 convert (4 elems / thread)
// ---------------------------------------------------------------------------
__global__ void cvt_f32_f16(const float* __restrict__ src, _Float16* __restrict__ dst, int n) {
    int i = (blockIdx.x * blockDim.x + threadIdx.x) * 4;
    if (i + 3 < n) {
        float4 v = *(const float4*)(src + i);
        dst[i + 0] = (_Float16)v.x;
        dst[i + 1] = (_Float16)v.y;
        dst[i + 2] = (_Float16)v.z;
        dst[i + 3] = (_Float16)v.w;
    }
}

// ---------------------------------------------------------------------------
// Projection GEMM: Y = X @ W^T (X: [B*S, D] f16, W: [D_out, D_in] f16).
// All 8 waves of a block share one 16-row X tile: it is staged into LDS once
// via GLOBAL_LOAD_ASYNC_TO_LDS_B128 (ASYNCcnt + s_wait_asynccnt + barrier),
// then each wave computes a 16x64 output strip (ng = wave id), reusing each
// LDS A-fragment 4x. K = 512 in 16 WMMA(16x16x32) steps per tile.
// p = blockIdx.y: 0 -> Q (scaled by 1/sqrt(D)), 1 -> K, 2 -> V stored
// TRANSPOSED as Vt[b][d][s] so attention's P@V B-fragments are contiguous.
// ---------------------------------------------------------------------------
__global__ __launch_bounds__(256)
void proj_kernel(const _Float16* __restrict__ X, const _Float16* __restrict__ W,
                 _Float16* __restrict__ Q, _Float16* __restrict__ K,
                 _Float16* __restrict__ Vt) {
    __shared__ _Float16 Xlds[16 * DD];   // 16 KB shared X tile

    const int lane = threadIdx.x & 31;
    const int ng   = threadIdx.x >> 5;   // wave id = 64-col group 0..7
    const int p    = blockIdx.y;
    const int mt   = blockIdx.x;         // 0 .. 1023 (16-row tile)
    const int m    = lane & 15;
    const int kh   = lane >> 4;

    // ---- cooperative async copy: X tile (1024 x 16B chunks) -> LDS ----
    {
        const _Float16* xbase = X + (size_t)mt * 16 * DD;
#pragma unroll
        for (int c = 0; c < 4; ++c) {
            int chunk = threadIdx.x + c * 256;                 // 0..1023
            uint32_t ldsoff = (uint32_t)(uintptr_t)&Xlds[chunk * 8];
            const void* g = (const void*)(xbase + chunk * 8);
            asm volatile("global_load_async_to_lds_b128 %0, %1, off"
                         :: "v"(ldsoff), "v"(g) : "memory");
        }
        asm volatile("s_wait_asynccnt 0x0" ::: "memory");
        __syncthreads();
    }

    const _Float16* Wp = W + (size_t)p * DD * DD;
    const _Float16* brow[4];
#pragma unroll
    for (int tt = 0; tt < 4; ++tt)
        brow[tt] = Wp + (size_t)(ng * 64 + tt * 16 + m) * DD;

    const _Float16* arow = &Xlds[m * DD];

    v8f acc[4] = {v8f{}, v8f{}, v8f{}, v8f{}};
#pragma unroll
    for (int k = 0; k < DD; k += 32) {
        half8 alo = *(const half8*)(arow + k + 16 * kh);        // ds_load_b128
        half8 ahi = *(const half8*)(arow + k + 16 * kh + 8);
        v16h a = SHUF16(alo, ahi);
#pragma unroll
        for (int tt = 0; tt < 4; ++tt) {
            half8 blo = *(const half8*)(brow[tt] + k + 16 * kh);
            half8 bhi = *(const half8*)(brow[tt] + k + 16 * kh + 8);
            v16h bfrag = SHUF16(blo, bhi);
            acc[tt] = __builtin_amdgcn_wmma_f32_16x16x32_f16(false, a, false, bfrag,
                                                             (short)0, acc[tt], false, false);
        }
    }

    if (p == 2) {
        // V: transposed store Vt[b][d][s]; each lane owns 8 consecutive rows
        // (s values) of one column -> one packed 16B store per tile.
        const int gr0  = mt * 16 + 8 * kh;     // first of 8 consecutive rows
        const int bidx = gr0 >> 12;            // / 4096 (tile never crosses batch)
        const int s0   = gr0 & (SS - 1);
#pragma unroll
        for (int tt = 0; tt < 4; ++tt) {
            const int col = ng * 64 + tt * 16 + m;
            half8 hv;
#pragma unroll
            for (int r = 0; r < 8; ++r) hv[r] = (_Float16)acc[tt][r];
            *(half8*)(Vt + (size_t)bidx * DD * SS + (size_t)col * SS + s0) = hv;
        }
    } else {
        const float scale = (p == 0) ? 0.04419417382415922f : 1.0f;  // 1/sqrt(512)
        _Float16* Y = (p == 0) ? Q : K;
#pragma unroll
        for (int tt = 0; tt < 4; ++tt)
#pragma unroll
            for (int r = 0; r < 8; ++r) {
                int row = mt * 16 + r + 8 * kh;
                int col = ng * 64 + tt * 16 + m;
                Y[(size_t)row * DD + col] = (_Float16)(acc[tt][r] * scale);
            }
    }
}

// ---------------------------------------------------------------------------
// Flash attention: one workgroup (8 waves) per (batch, 16-row query tile).
// Q tile register-resident per wave. Per 128-key block: wave w computes its
// 16x16 score tile (QK^T), waves merge softmax stats via LDS, P (f16) staged
// in a shared 16x128 LDS tile, then wave w accumulates output columns
// [64w, 64w+64) with P@V WMMAs reading the transposed Vt (contiguous frags).
// K/Vt fragments have zero intra-wave reuse -> direct global_load_b128 path.
// ---------------------------------------------------------------------------
__global__ __launch_bounds__(256)
void attn_kernel(const _Float16* __restrict__ Qg, const _Float16* __restrict__ Kg,
                 const _Float16* __restrict__ Vt, float* __restrict__ Out) {
    __shared__ _Float16 Plds[16][128];
    __shared__ float blockmax[8][16];
    __shared__ float blocksum[8][16];

    const int lane = threadIdx.x & 31;
    const int w    = threadIdx.x >> 5;
    const int m    = lane & 15;
    const int kh   = lane >> 4;
    const int mt   = blockIdx.x;   // query row tile 0..255
    const int b    = blockIdx.y;   // batch

    const size_t boff = (size_t)b * SS * DD;
    const _Float16* vtb = Vt + (size_t)b * DD * SS;   // Vt[b][d][s]

    // Q fragments: 16 rows x 512, A-matrix layout, 16 chunks of K=32.
    const _Float16* qrow = Qg + boff + (size_t)(mt * 16 + m) * DD;
    v16h qa[16];
#pragma unroll
    for (int kk = 0; kk < 16; ++kk) {
        half8 lo = *(const half8*)(qrow + kk * 32 + 16 * kh);
        half8 hi = *(const half8*)(qrow + kk * 32 + 16 * kh + 8);
        qa[kk] = SHUF16(lo, hi);
    }

    v8f acc[4] = {v8f{}, v8f{}, v8f{}, v8f{}};   // 16x64 output slice
    float mrow[8], lrow[8];
#pragma unroll
    for (int r = 0; r < 8; ++r) { mrow[r] = -1e30f; lrow[r] = 0.0f; }

    for (int jb = 0; jb < SS / 128; ++jb) {
        // prefetch next block's K rows / Vt columns (global_prefetch_b8)
        if (jb + 1 < SS / 128) {
            __builtin_prefetch(Kg + boff + (size_t)((jb + 1) * 128 + w * 16 + m) * DD, 0, 0);
            __builtin_prefetch(vtb + (size_t)(w * 64 + m) * SS + (jb + 1) * 128, 0, 0);
        }

        // ----- S = Q K^T for this wave's 16-key tile -----
        const int keybase = jb * 128 + w * 16;
        const _Float16* krow = Kg + boff + (size_t)(keybase + m) * DD;
        v8f s = {};
#pragma unroll
        for (int kk = 0; kk < 16; ++kk) {
            half8 blo = *(const half8*)(krow + kk * 32 + 16 * kh);
            half8 bhi = *(const half8*)(krow + kk * 32 + 16 * kh + 8);
            v16h kb = SHUF16(blo, bhi);
            s = __builtin_amdgcn_wmma_f32_16x16x32_f16(false, qa[kk], false, kb,
                                                       (short)0, s, false, false);
        }

        // ----- per-row tile max (reduce across 16 lanes) -----
        float tmax[8];
#pragma unroll
        for (int r = 0; r < 8; ++r) {
            float t = s[r];
#pragma unroll
            for (int off = 8; off >= 1; off >>= 1) t = fmaxf(t, __shfl_xor(t, off, 16));
            tmax[r] = t;
        }
        if (m == 0) {
#pragma unroll
            for (int r = 0; r < 8; ++r) blockmax[w][r + 8 * kh] = tmax[r];
        }
        __syncthreads();

        // ----- merge maxima across waves; rescale factors -----
        float fac[8], mnew[8];
#pragma unroll
        for (int r = 0; r < 8; ++r) {
            float bm = mrow[r];
#pragma unroll
            for (int w2 = 0; w2 < 8; ++w2) bm = fmaxf(bm, blockmax[w2][r + 8 * kh]);
            mnew[r] = bm;
            fac[r]  = __expf(mrow[r] - bm);
            mrow[r] = bm;
        }

        // ----- P = exp(S - m_new); stage in LDS; per-row tile sums -----
#pragma unroll
        for (int r = 0; r < 8; ++r) {
            float pv = __expf(s[r] - mnew[r]);
            Plds[r + 8 * kh][w * 16 + m] = (_Float16)pv;
            float t = pv;
#pragma unroll
            for (int off = 8; off >= 1; off >>= 1) t += __shfl_xor(t, off, 16);
            if (m == 0) blocksum[w][r + 8 * kh] = t;
        }

        // ----- rescale output accumulators -----
#pragma unroll
        for (int tt = 0; tt < 4; ++tt)
#pragma unroll
            for (int r = 0; r < 8; ++r) acc[tt][r] *= fac[r];

        __syncthreads();

        // ----- update running row sums -----
#pragma unroll
        for (int r = 0; r < 8; ++r) {
            float su = 0.0f;
#pragma unroll
            for (int w2 = 0; w2 < 8; ++w2) su += blocksum[w2][r + 8 * kh];
            lrow[r] = lrow[r] * fac[r] + su;
        }

        // ----- O += P @ V : K = 128 keys in 4 chunks of 32; B-fragments are
        //       contiguous 16-key runs of one Vt column (2x b128 loads) -----
#pragma unroll
        for (int kc = 0; kc < 4; ++kc) {
            half8 plo = *(const half8*)(&Plds[m][kc * 32 + 16 * kh]);
            half8 phi = *(const half8*)(&Plds[m][kc * 32 + 16 * kh + 8]);
            v16h pa = SHUF16(plo, phi);
#pragma unroll
            for (int tt = 0; tt < 4; ++tt) {
                const _Float16* vp = vtb + (size_t)(w * 64 + tt * 16 + m) * SS
                                         + jb * 128 + kc * 32 + 16 * kh;
                half8 vlo = *(const half8*)(vp);
                half8 vhi = *(const half8*)(vp + 8);
                v16h vb = SHUF16(vlo, vhi);
                acc[tt] = __builtin_amdgcn_wmma_f32_16x16x32_f16(false, pa, false, vb,
                                                                 (short)0, acc[tt], false, false);
            }
        }
        __syncthreads();  // protect Plds / blockmax / blocksum reuse
    }

    // ----- final normalize + store f32 output -----
#pragma unroll
    for (int tt = 0; tt < 4; ++tt)
#pragma unroll
        for (int r = 0; r < 8; ++r) {
            size_t idx = boff + (size_t)(mt * 16 + r + 8 * kh) * DD + w * 64 + tt * 16 + m;
            Out[idx] = acc[tt][r] / lrow[r];
        }
}

// ---------------------------------------------------------------------------
extern "C" void kernel_launch(void* const* d_in, const int* in_sizes, int n_in,
                              void* d_out, int out_size, void* d_ws, size_t ws_size,
                              hipStream_t stream) {
    const float* token = (const float*)d_in[0];
    const float* W1    = (const float*)d_in[1];
    const float* W2    = (const float*)d_in[2];
    const float* W3    = (const float*)d_in[3];
    float*       out   = (float*)d_out;

    const size_t nTok = (size_t)BB * SS * DD;     // 8,388,608
    const size_t nW   = (size_t)DD * DD;          // 262,144

    _Float16* X16  = (_Float16*)d_ws;
    _Float16* W16  = X16 + nTok;
    _Float16* Q16  = W16 + 3 * nW;
    _Float16* K16  = Q16 + nTok;
    _Float16* Vt16 = K16 + nTok;                  // transposed V: [B][D][S]

    // 1) f32 -> f16 converts
    {
        int n = (int)nTok;
        int threads = n / 4;
        cvt_f32_f16<<<(threads + 255) / 256, 256, 0, stream>>>(token, X16, n);
        int nw = (int)nW, tw = nw / 4;
        cvt_f32_f16<<<(tw + 255) / 256, 256, 0, stream>>>(W1, W16 + 0 * nW, nw);
        cvt_f32_f16<<<(tw + 255) / 256, 256, 0, stream>>>(W2, W16 + 1 * nW, nw);
        cvt_f32_f16<<<(tw + 255) / 256, 256, 0, stream>>>(W3, W16 + 2 * nW, nw);
    }

    // 2) Q/K/Vt projections: one block per 16-row tile, 8 waves = 8 col groups
    proj_kernel<<<dim3(1024, 3), 256, 0, stream>>>(X16, W16, Q16, K16, Vt16);

    // 3) flash attention: (query-tile, batch) grid
    attn_kernel<<<dim3(SS / 16, BB), 256, 0, stream>>>(Q16, K16, Vt16, out);
}
